// ConvSS2D_35596688949273
// MI455X (gfx1250) — matrix operand order repaired and checked by reading, not compiled, not run on hardware
//
#include <hip/hip_runtime.h>
#include <math.h>

typedef __attribute__((ext_vector_type(16))) _Float16 v16h;
typedef __attribute__((ext_vector_type(8)))  float    v8f;

#define L_TOT 8192
#define H_DIM 64
#define HSHIFT 6
#define W_DIM 128
#define DIN   128
#define HD    128
#define IDIM  192
#define NSTATE 16
#define RRANK 8
#define KDIR  4
#define MPROJ (RRANK + 2*NSTATE)   // 40

union V16H { v16h v; _Float16 h[16]; };
union V8F  { v8f v;  float    f[8];  };

// ---- CDNA5 async global->LDS path (asm; wait via confirmed builtin) --------
#if defined(__has_builtin)
#if __has_builtin(__builtin_amdgcn_s_wait_asynccnt)
#define HAVE_ASYNC_LDS 1
#endif
#endif

// Stage a 32x128 f32 B-panel (rows stride L_TOT) into LDS. 256 threads.
__device__ __forceinline__ void stage_b_chunk(const float* __restrict__ src,
                                              float* buf, int tid)
{
  int r = tid >> 3;            // 0..31 rows
  int c = (tid & 7) << 4;      // 16-float column slice
  const float* g = src + r * L_TOT + c;
  float* l = buf + r * 128 + c;
#ifdef HAVE_ASYNC_LDS
  #pragma unroll
  for (int j = 0; j < 4; ++j) {
    unsigned lds_addr = (unsigned)(unsigned long long)(l + 4 * j); // addr[31:0] = LDS offset
    unsigned long long gaddr = (unsigned long long)(g + 4 * j);
    asm volatile("global_load_async_to_lds_b128 %0, %1, off"
                 :: "v"(lds_addr), "v"(gaddr)
                 : "memory");
  }
#else
  #pragma unroll
  for (int j = 0; j < 4; ++j)
    *(float4*)(l + 4 * j) = *(const float4*)(g + 4 * j);
#endif
}

__device__ __forceinline__ void wait_stage() {
#ifdef HAVE_ASYNC_LDS
  __builtin_amdgcn_s_wait_asynccnt(0);
#endif
  __syncthreads();
}

// Per-lane K index for the 16-bit A operand (16x32), contiguous k = 16*khalf + i (matches B).
__device__ __forceinline__ int a_k_index(int khalf, int i) {
  return khalf * 16 + i;
}

// cross_scan / cross_merge canonical-position map.
__device__ __forceinline__ int src_index(int k, int l) {
  switch (k & 3) {
    case 0: return l;
    case 1: return ((l & (H_DIM - 1)) << 7) + (l >> HSHIFT);          // h*W + w
    case 2: return L_TOT - 1 - l;
    default: { int l2 = L_TOT - 1 - l;
               return ((l2 & (H_DIM - 1)) << 7) + (l2 >> HSHIFT); }
  }
}

// ---------------- GEMM 1: hx = w_proj (128x320) * [h;x] (320xL) + b_proj ----
__global__ __launch_bounds__(256) void k_gemm_in(
    const float* __restrict__ hsrc, const float* __restrict__ xsrc,
    const float* __restrict__ w, const float* __restrict__ bias,
    float* __restrict__ out)
{
  const int K = HD + IDIM, NC = K / 32;          // 320, 10 chunks
  __shared__ float bbuf[2][32 * 128];
  int tid = threadIdx.x;
  int wave = tid >> 5, lane = tid & 31;
  int khalf = lane >> 4, lan = lane & 15;
  int col0 = blockIdx.x * 128;
  int col = col0 + wave * 16 + lan;
  int m0 = blockIdx.y * 16;

  stage_b_chunk(hsrc + col0, bbuf[0], tid);      // chunk 0 rows 0..31 (< HD)
  V8F acc;
  #pragma unroll
  for (int r = 0; r < 8; ++r) acc.f[r] = 0.f;
  for (int kc = 0; kc < NC; ++kc) {
    wait_stage();                                 // buf[kc&1] ready on all waves
    if (kc + 1 < NC) {
      int k0 = (kc + 1) * 32;
      const float* src = (k0 < HD) ? (hsrc + (size_t)k0 * L_TOT + col0)
                                   : (xsrc + (size_t)(k0 - HD) * L_TOT + col0);
      stage_b_chunk(src, bbuf[(kc + 1) & 1], tid);
    }
    const float* bb = bbuf[kc & 1];
    V16H a, b;
    const float* arow = w + (m0 + lan) * K + kc * 32;
    #pragma unroll
    for (int i = 0; i < 16; ++i) a.h[i] = (_Float16)arow[a_k_index(khalf, i)];
    #pragma unroll
    for (int i = 0; i < 16; ++i)
      b.h[i] = (_Float16)bb[(khalf * 16 + i) * 128 + wave * 16 + lan];
    acc.v = __builtin_amdgcn_wmma_f32_16x16x32_f16(false, a.v, false, b.v,
                                                   (short)0, acc.v, false, false);
  }
  #pragma unroll
  for (int r = 0; r < 8; ++r) {
    int m = m0 + khalf * 8 + r;
    out[m * L_TOT + col] = acc.f[r] + bias[m];
  }
}

// ---------------- GEMM 2: xz = w_in (256x128) * hx (128xL) ------------------
__global__ __launch_bounds__(256) void k_gemm_xz(
    const float* __restrict__ w, const float* __restrict__ hx,
    float* __restrict__ xz)
{
  const int K = HD, NC = K / 32;                 // 128, 4 chunks
  __shared__ float bbuf[2][32 * 128];
  int tid = threadIdx.x;
  int wave = tid >> 5, lane = tid & 31;
  int khalf = lane >> 4, lan = lane & 15;
  int col0 = blockIdx.x * 128;
  int col = col0 + wave * 16 + lan;
  int m0 = blockIdx.y * 16;

  stage_b_chunk(hx + col0, bbuf[0], tid);
  V8F acc;
  #pragma unroll
  for (int r = 0; r < 8; ++r) acc.f[r] = 0.f;
  for (int kc = 0; kc < NC; ++kc) {
    wait_stage();
    if (kc + 1 < NC)
      stage_b_chunk(hx + (size_t)(kc + 1) * 32 * L_TOT + col0, bbuf[(kc + 1) & 1], tid);
    const float* bb = bbuf[kc & 1];
    V16H a, b;
    const float* arow = w + (m0 + lan) * K + kc * 32;
    #pragma unroll
    for (int i = 0; i < 16; ++i) a.h[i] = (_Float16)arow[a_k_index(khalf, i)];
    #pragma unroll
    for (int i = 0; i < 16; ++i)
      b.h[i] = (_Float16)bb[(khalf * 16 + i) * 128 + wave * 16 + lan];
    acc.v = __builtin_amdgcn_wmma_f32_16x16x32_f16(false, a.v, false, b.v,
                                                   (short)0, acc.v, false, false);
  }
  #pragma unroll
  for (int r = 0; r < 8; ++r)
    xz[(m0 + khalf * 8 + r) * L_TOT + col] = acc.f[r];
}

// ---------------- depthwise 3x3 conv (SAME) + bias + SiLU -------------------
__global__ __launch_bounds__(256) void k_conv(
    const float* __restrict__ xp, const float* __restrict__ wconv,
    const float* __restrict__ bconv, float* __restrict__ xconv)
{
  int idx = blockIdx.x * 256 + threadIdx.x;
  if (idx >= DIN * L_TOT) return;
  int d = idx >> 13;
  int p = idx & (L_TOT - 1);
  int hh = p >> 7, ww = p & (W_DIM - 1);
  const float* wc = wconv + d * 9;
  float acc = bconv[d];
  #pragma unroll
  for (int dy = -1; dy <= 1; ++dy)
    #pragma unroll
    for (int dx = -1; dx <= 1; ++dx) {
      int yy = hh + dy, xx = ww + dx;
      if (yy >= 0 && yy < H_DIM && xx >= 0 && xx < W_DIM)
        acc += wc[(dy + 1) * 3 + (dx + 1)] * xp[d * L_TOT + yy * W_DIM + xx];
    }
  float sig = 1.f / (1.f + __expf(-acc));
  xconv[idx] = acc * sig;
}

// ---------------- GEMM 3: dbc[k] = x_proj_w[k] (40x128) * xs[k] (128xL) -----
// Columns are permuted by the cross-scan map, so B is gathered directly.
__global__ __launch_bounds__(256) void k_gemm_dbc(
    const float* __restrict__ xpw, const float* __restrict__ xconv,
    float* __restrict__ dbc)
{
  const int K = DIN;
  int kdir = blockIdx.z;
  const float* A = xpw + kdir * MPROJ * K;
  float* out = dbc + kdir * MPROJ * L_TOT;
  int wave = threadIdx.x >> 5, lane = threadIdx.x & 31;
  int khalf = lane >> 4, lan = lane & 15;
  int colL = (blockIdx.x * 8 + wave) * 16 + lan;
  int m0 = blockIdx.y * 16;
  int p = src_index(kdir, colL);
  V8F acc;
  #pragma unroll
  for (int r = 0; r < 8; ++r) acc.f[r] = 0.f;
  for (int k0 = 0; k0 < K; k0 += 32) {
    V16H a, b;
    int mrow = m0 + lan;
    #pragma unroll
    for (int i = 0; i < 16; ++i)
      a.h[i] = (mrow < MPROJ) ? (_Float16)A[mrow * K + k0 + a_k_index(khalf, i)]
                              : (_Float16)0.f;
    #pragma unroll
    for (int i = 0; i < 16; ++i)
      b.h[i] = (_Float16)xconv[(k0 + khalf * 16 + i) * L_TOT + p];
    acc.v = __builtin_amdgcn_wmma_f32_16x16x32_f16(false, a.v, false, b.v,
                                                   (short)0, acc.v, false, false);
  }
  #pragma unroll
  for (int r = 0; r < 8; ++r) {
    int m = m0 + khalf * 8 + r;
    if (m < MPROJ) out[m * L_TOT + colL] = acc.f[r];
  }
}

// ---------------- selective scan: one workgroup per (k,d) -------------------
#define SCAN_T 256
#define CHUNK (L_TOT / SCAN_T)     // 32
__global__ __launch_bounds__(SCAN_T) void k_scan(
    const float* __restrict__ xconv, const float* __restrict__ dbc,
    const float* __restrict__ dt_w, const float* __restrict__ dt_b,
    const float* __restrict__ A_log, const float* __restrict__ D_ssm,
    float* __restrict__ ysb)
{
  int kd = blockIdx.x;
  int k = kd >> 7, d = kd & (DIN - 1);
  int tid = threadIdx.x;
  const float* dbcK = dbc + (size_t)k * MPROJ * L_TOT;
  const float* xu = xconv + (size_t)d * L_TOT;

  float dtw[RRANK];
  #pragma unroll
  for (int r = 0; r < RRANK; ++r) dtw[r] = dt_w[(k * DIN + d) * RRANK + r];
  float dtb = dt_b[k * DIN + d];
  float An[NSTATE];
  #pragma unroll
  for (int n = 0; n < NSTATE; ++n) An[n] = -__expf(A_log[(k * DIN + d) * NSTATE + n]);
  float Dv = D_ssm[k * DIN + d];

  __shared__ float sA[SCAN_T][NSTATE + 1];
  __shared__ float sB[SCAN_T][NSTATE + 1];

  float Ap[NSTATE], Bc[NSTATE];
  #pragma unroll
  for (int n = 0; n < NSTATE; ++n) { Ap[n] = 1.f; Bc[n] = 0.f; }
  int l0 = tid * CHUNK;
  for (int i = 0; i < CHUNK; ++i) {
    int l = l0 + i;
    int p = src_index(k, l);
    __builtin_prefetch(&dbcK[RRANK * L_TOT + l + 1], 0, 0);
    float u = xu[p];
    float s = dtb;
    #pragma unroll
    for (int r = 0; r < RRANK; ++r) s += dtw[r] * dbcK[r * L_TOT + l];
    float dt = (s > 20.f) ? s : __logf(1.f + __expf(s));   // softplus
    float du = dt * u;
    #pragma unroll
    for (int n = 0; n < NSTATE; ++n) {
      float a = __expf(dt * An[n]);
      float b = du * dbcK[(RRANK + n) * L_TOT + l];
      Ap[n] *= a;
      Bc[n] = a * Bc[n] + b;
    }
  }
  #pragma unroll
  for (int n = 0; n < NSTATE; ++n) { sA[tid][n] = Ap[n]; sB[tid][n] = Bc[n]; }
  __syncthreads();

  for (int off = 1; off < SCAN_T; off <<= 1) {
    float pa[NSTATE], pb[NSTATE];
    if (tid >= off) {
      #pragma unroll
      for (int n = 0; n < NSTATE; ++n) { pa[n] = sA[tid - off][n]; pb[n] = sB[tid - off][n]; }
    }
    __syncthreads();
    if (tid >= off) {
      #pragma unroll
      for (int n = 0; n < NSTATE; ++n) {
        float aC = sA[tid][n], bC = sB[tid][n];
        sA[tid][n] = pa[n] * aC;
        sB[tid][n] = aC * pb[n] + bC;
      }
    }
    __syncthreads();
  }

  float hs[NSTATE];
  #pragma unroll
  for (int n = 0; n < NSTATE; ++n) hs[n] = (tid == 0) ? 0.f : sB[tid - 1][n];

  float* yout = ysb + (size_t)(k * DIN + d) * L_TOT;
  for (int i = 0; i < CHUNK; ++i) {
    int l = l0 + i;
    int p = src_index(k, l);
    float u = xu[p];
    float s = dtb;
    #pragma unroll
    for (int r = 0; r < RRANK; ++r) s += dtw[r] * dbcK[r * L_TOT + l];
    float dt = (s > 20.f) ? s : __logf(1.f + __expf(s));
    float du = dt * u;
    float y = Dv * u;
    #pragma unroll
    for (int n = 0; n < NSTATE; ++n) {
      float a = __expf(dt * An[n]);
      float b = du * dbcK[(RRANK + n) * L_TOT + l];
      hs[n] = a * hs[n] + b;
      y += dbcK[(RRANK + NSTATE + n) * L_TOT + l] * hs[n];
    }
    yout[p] = y;
  }
}

// ---------------- merge 4 dirs + LayerNorm(ch) + SiLU gate ------------------
__global__ __launch_bounds__(128) void k_merge(
    const float* __restrict__ ysb, const float* __restrict__ z,
    const float* __restrict__ ln_g, const float* __restrict__ ln_b,
    float* __restrict__ ymod)
{
  int p = blockIdx.x;
  int d = threadIdx.x;
  float v = 0.f;
  #pragma unroll
  for (int k = 0; k < KDIR; ++k) v += ysb[(size_t)(k * DIN + d) * L_TOT + p];
  __shared__ float s1[DIN], s2[DIN];
  s1[d] = v; s2[d] = v * v;
  __syncthreads();
  for (int off = DIN / 2; off > 0; off >>= 1) {
    if (d < off) { s1[d] += s1[d + off]; s2[d] += s2[d + off]; }
    __syncthreads();
  }
  float mu = s1[0] * (1.f / DIN);
  float var = s2[0] * (1.f / DIN) - mu * mu;
  float yn = (v - mu) * rsqrtf(var + 1e-5f);
  yn = yn * ln_g[d] + ln_b[d];
  float zv = z[d * L_TOT + p];
  float sig = 1.f / (1.f + __expf(-zv));
  ymod[d * L_TOT + p] = yn * (zv * sig);
}

// ---------------- GEMM 4: out = h + tanh(w_out (128x128) * ymod) ------------
__global__ __launch_bounds__(256) void k_gemm_out(
    const float* __restrict__ w, const float* __restrict__ ymod,
    const float* __restrict__ hin, float* __restrict__ out)
{
  const int K = DIN, NC = K / 32;
  __shared__ float bbuf[2][32 * 128];
  int tid = threadIdx.x;
  int wave = tid >> 5, lane = tid & 31;
  int khalf = lane >> 4, lan = lane & 15;
  int col0 = blockIdx.x * 128;
  int col = col0 + wave * 16 + lan;
  int m0 = blockIdx.y * 16;

  stage_b_chunk(ymod + col0, bbuf[0], tid);
  V8F acc;
  #pragma unroll
  for (int r = 0; r < 8; ++r) acc.f[r] = 0.f;
  for (int kc = 0; kc < NC; ++kc) {
    wait_stage();
    if (kc + 1 < NC)
      stage_b_chunk(ymod + (size_t)(kc + 1) * 32 * L_TOT + col0, bbuf[(kc + 1) & 1], tid);
    const float* bb = bbuf[kc & 1];
    V16H a, b;
    const float* arow = w + (m0 + lan) * K + kc * 32;
    #pragma unroll
    for (int i = 0; i < 16; ++i) a.h[i] = (_Float16)arow[a_k_index(khalf, i)];
    #pragma unroll
    for (int i = 0; i < 16; ++i)
      b.h[i] = (_Float16)bb[(khalf * 16 + i) * 128 + wave * 16 + lan];
    acc.v = __builtin_amdgcn_wmma_f32_16x16x32_f16(false, a.v, false, b.v,
                                                   (short)0, acc.v, false, false);
  }
  #pragma unroll
  for (int r = 0; r < 8; ++r) {
    int m = m0 + khalf * 8 + r;
    out[m * L_TOT + col] = hin[m * L_TOT + col] + tanhf(acc.f[r]);
  }
}

extern "C" void kernel_launch(void* const* d_in, const int* in_sizes, int n_in,
                              void* d_out, int out_size, void* d_ws, size_t ws_size,
                              hipStream_t stream) {
  (void)in_sizes; (void)n_in; (void)out_size; (void)ws_size;
  const float* h       = (const float*)d_in[0];
  const float* x       = (const float*)d_in[1];
  const float* w_proj  = (const float*)d_in[2];
  const float* b_proj  = (const float*)d_in[3];
  const float* w_in    = (const float*)d_in[4];
  const float* w_conv  = (const float*)d_in[5];
  const float* b_conv  = (const float*)d_in[6];
  const float* x_projw = (const float*)d_in[7];
  const float* dt_w    = (const float*)d_in[8];
  const float* dt_b    = (const float*)d_in[9];
  const float* A_log   = (const float*)d_in[10];
  const float* D_ssm   = (const float*)d_in[11];
  const float* ln_g    = (const float*)d_in[12];
  const float* ln_b    = (const float*)d_in[13];
  const float* w_out   = (const float*)d_in[14];

  float* ws    = (float*)d_ws;
  float* hx    = ws;                            // 128*8192
  float* xz    = hx + DIN * L_TOT;              // 256*8192 (xp rows 0-127, z rows 128-255)
  float* xconv = xz + 2 * DIN * L_TOT;          // 128*8192
  float* dbc   = xconv + DIN * L_TOT;           // 4*40*8192
  float* ysb   = dbc + KDIR * MPROJ * L_TOT;    // 4*128*8192
  float* ymod  = ysb + KDIR * DIN * L_TOT;      // 128*8192

  dim3 blk(256);
  k_gemm_in <<<dim3(64, HD / 16),      blk, 0, stream>>>(h, x, w_proj, b_proj, hx);
  k_gemm_xz <<<dim3(64, 2 * DIN / 16), blk, 0, stream>>>(w_in, hx, xz);
  k_conv    <<<dim3(DIN * L_TOT / 256), blk, 0, stream>>>(xz, w_conv, b_conv, xconv);
  k_gemm_dbc<<<dim3(64, 3, KDIR),      blk, 0, stream>>>(x_projw, xconv, dbc);
  k_scan    <<<dim3(KDIR * DIN),       blk, 0, stream>>>(xconv, dbc, dt_w, dt_b, A_log, D_ssm, ysb);
  k_merge   <<<dim3(L_TOT), dim3(DIN), 0, stream>>>(ysb, xz + DIN * L_TOT, ln_g, ln_b, ymod);
  k_gemm_out<<<dim3(64, HD / 16),      blk, 0, stream>>>(w_out, ymod, h, (float*)d_out);
}
